// MultiHeadAttention_62466004353308
// MI455X (gfx1250) — matrix operand order repaired and checked by reading, not compiled, not run on hardware
//
#include <hip/hip_runtime.h>

// ---------------------------------------------------------------------------
// MHA forward for MI455X (gfx1250, wave32, WMMA 16x16x32 bf16 / f32-acc).
// Round 2: software-pipelined GEMM loops (next-chunk operands preloaded into
// distinct registers so the backend can issue partial s_wait_loadcnt instead
// of serializing every WMMA on s_wait_loadcnt 0), batched K+V load clauses in
// the flash-attention inner loop so V loads hide under QK^T WMMA + softmax.
// ---------------------------------------------------------------------------

#define S_LEN   2048
#define D_MODEL 1024
#define N_HEADS 16
#define D_K     64
#define BATCH   2
#define M_ROWS  (BATCH * S_LEN)      // 4096

typedef __attribute__((ext_vector_type(16))) __bf16 v16bf;
typedef __attribute__((ext_vector_type(8)))  float  v8f;

union Frag {
    v16bf        v;
    uint4        u[2];
    unsigned int w[8];
};

static __device__ __forceinline__ v8f vzero8() {
    v8f z = {0.f, 0.f, 0.f, 0.f, 0.f, 0.f, 0.f, 0.f};
    return z;
}

static __device__ __forceinline__ v8f wmma_bf16(v16bf a, v16bf b, v8f c) {
    return __builtin_amdgcn_wmma_f32_16x16x32_bf16(
        /*neg_a=*/false, a, /*neg_b=*/false, b,
        /*c_mod=*/(short)0, c, /*reuse_a=*/false, /*reuse_b=*/false);
}

static __device__ __forceinline__ unsigned short to_bf16(float f) {
    union { float f; unsigned u; } cv; cv.f = f;
    unsigned u = cv.u;
    u += 0x7FFFu + ((u >> 16) & 1u);   // round-to-nearest-even
    return (unsigned short)(u >> 16);
}

// A-operand (16x32, 16-bit): lane holds row (m = lane&15);
// low half-wave: K = {0..7, 16..23} (hi half: +8). Two b128 at +0 / +32B.
static __device__ __forceinline__ v16bf load_a(const unsigned short* p) {
    Frag f;
    f.u[0] = *(const uint4*)(p);
    f.u[1] = *(const uint4*)(p + 16);
    return f.v;
}

// B-operand (32x16, 16-bit): lane holds column (n = lane&15);
// low half-wave: K = {0..7, 16..23} (hi half: +8). Two b128 at +0 / +32B.
static __device__ __forceinline__ v16bf load_b(const unsigned short* p) {
    Frag f;
    f.u[0] = *(const uint4*)(p);
    f.u[1] = *(const uint4*)(p + 16);
    return f.v;
}

// ---------------------------------------------------------------------------
// fp32 -> bf16 conversion, 4 elements per thread
// ---------------------------------------------------------------------------
__global__ void mha_cvt_bf16_x4(const float* __restrict__ in,
                                unsigned short* __restrict__ out, int n4) {
    int i = blockIdx.x * blockDim.x + threadIdx.x;
    if (i < n4) {
        float4 f = ((const float4*)in)[i];
        ushort4 o;
        o.x = to_bf16(f.x); o.y = to_bf16(f.y);
        o.z = to_bf16(f.z); o.w = to_bf16(f.w);
        ((ushort4*)out)[i] = o;
    }
}

// ---------------------------------------------------------------------------
// QKV projection: y = x @ W^T + b  (M=4096, N=1024, K=1024)
// One wave -> 16 rows x 64 cols (one head). Software-pipelined over k-chunks.
// which (blockIdx.z): 0=Q (scaled 1/8, [bh][s][d]), 1=K ([bh][s][d]),
//                     2=V (transposed, [bh][d][s]).
// ---------------------------------------------------------------------------
__global__ __launch_bounds__(128) void mha_qkv_kernel(
    const unsigned short* __restrict__ xb,
    const unsigned short* __restrict__ wq,
    const unsigned short* __restrict__ wk,
    const unsigned short* __restrict__ wv,
    const float* __restrict__ bq, const float* __restrict__ bk,
    const float* __restrict__ bv,
    unsigned short* __restrict__ Qb, unsigned short* __restrict__ Kb,
    unsigned short* __restrict__ Vt) {
    const int lane = threadIdx.x & 31;
    const int wid  = threadIdx.x >> 5;
    const int hi   = lane >> 4;
    const int ln   = lane & 15;
    const int koffA = hi ? 8 : 0;
    const int koffB = hi ? 8 : 0;

    const int which = blockIdx.z;
    const unsigned short* W = (which == 0) ? wq : ((which == 1) ? wk : wv);
    const float* bias       = (which == 0) ? bq : ((which == 1) ? bk : bv);

    const int mtile = blockIdx.x * 4 + wid;   // 0..255
    const int m0    = mtile * 16;
    const int h     = blockIdx.y;             // 0..15
    const int n0    = h * D_K;

    const unsigned short* arow = xb + (m0 + ln) * D_MODEL + koffA;
    const unsigned short* brow = W + (n0 + ln) * D_MODEL + koffB;

    v8f acc[4] = {vzero8(), vzero8(), vzero8(), vzero8()};

    // prologue: chunk 0 operands
    v16bf a  = load_a(arow);
    v16bf b0 = load_b(brow);
    v16bf b1 = load_b(brow + 16 * D_MODEL);
    v16bf b2 = load_b(brow + 32 * D_MODEL);
    v16bf b3 = load_b(brow + 48 * D_MODEL);

    for (int c = 0; c < D_MODEL / 32 - 1; ++c) {
        const int k1 = 32 * (c + 1);
        // issue next-chunk loads into fresh registers BEFORE consuming current
        v16bf an  = load_a(arow + k1);
        v16bf bn0 = load_b(brow + k1);
        v16bf bn1 = load_b(brow + 16 * D_MODEL + k1);
        v16bf bn2 = load_b(brow + 32 * D_MODEL + k1);
        v16bf bn3 = load_b(brow + 48 * D_MODEL + k1);
        acc[0] = wmma_bf16(a, b0, acc[0]);
        acc[1] = wmma_bf16(a, b1, acc[1]);
        acc[2] = wmma_bf16(a, b2, acc[2]);
        acc[3] = wmma_bf16(a, b3, acc[3]);
        a = an; b0 = bn0; b1 = bn1; b2 = bn2; b3 = bn3;
    }
    acc[0] = wmma_bf16(a, b0, acc[0]);
    acc[1] = wmma_bf16(a, b1, acc[1]);
    acc[2] = wmma_bf16(a, b2, acc[2]);
    acc[3] = wmma_bf16(a, b3, acc[3]);

    const int b  = m0 >> 11;          // / S_LEN
    const int s0 = m0 & (S_LEN - 1);
    const int bh = b * N_HEADS + h;
#pragma unroll
    for (int nt = 0; nt < 4; ++nt) {
        const int d   = nt * 16 + ln;
        const float bv_ = bias[n0 + d];
#pragma unroll
        for (int r = 0; r < 8; ++r) {
            const int s = s0 + r + (hi ? 8 : 0);
            float val = acc[nt][r] + bv_;
            if (which == 0) val *= 0.125f;                       // 1/sqrt(d_k)
            if (which == 2)
                Vt[(bh * D_K + d) * S_LEN + s] = to_bf16(val);   // transposed
            else {
                unsigned short* out = (which == 0) ? Qb : Kb;
                out[(bh * S_LEN + s) * D_K + d] = to_bf16(val);
            }
        }
    }
}

// ---------------------------------------------------------------------------
// Flash attention. One wave owns a 16-query strip of one (b,h).
// S^T = K*Q^T so D-layout (lane=query, vgpr=key) repacks directly into the
// A-operand of P*V without any LDS transpose. All 16 b128 loads of a key
// block are issued up front; V loads drain under QK^T WMMAs + softmax VALU.
// ---------------------------------------------------------------------------
__global__ __launch_bounds__(128) void mha_attn_kernel(
    const unsigned short* __restrict__ Qb, const unsigned short* __restrict__ Kb,
    const unsigned short* __restrict__ Vt, const int* __restrict__ am,
    unsigned short* __restrict__ ctx) {
    const int lane = threadIdx.x & 31;
    const int wid  = threadIdx.x >> 5;
    const int hi   = lane >> 4;
    const int ln   = lane & 15;
    const int koffA = hi ? 8 : 0;
    const int koffB = hi ? 8 : 0;

    const int bh = blockIdx.y;
    const int b  = bh >> 4;
    const int h  = bh & (N_HEADS - 1);
    const int qtile = blockIdx.x * 4 + wid;   // 0..127
    const int q0    = qtile * 16;
    const int qg    = q0 + ln;                // query owned in S^T layout

    // Q as B-operand, both 32-wide d chunks, held for the whole strip.
    Frag qf[2];
    {
        const unsigned short* qrow = Qb + (bh * S_LEN + q0 + ln) * D_K + koffB;
        qf[0].u[0] = *(const uint4*)(qrow);
        qf[0].u[1] = *(const uint4*)(qrow + 16);
        qf[1].u[0] = *(const uint4*)(qrow + 32);
        qf[1].u[1] = *(const uint4*)(qrow + 48);
    }

    const unsigned short* kbase = Kb + (bh * S_LEN + ln) * D_K + koffA;
    const unsigned short* vbase = Vt + (bh * D_K + ln) * S_LEN + koffB;

    v8f o[4] = {vzero8(), vzero8(), vzero8(), vzero8()};
    float mprev = -1.0e30f;
    float l     = 0.0f;

    const int kend = q0 + 15;
    for (int kb = 0; kb <= kend; kb += 32) {
        // per-key validity from attention_mask, one ballot for the 32-key block
        unsigned mask32 = (unsigned)__ballot(am[b * S_LEN + kb + lane] != 0);

        // ---- issue ALL loads for this block in one batch (in-order returns:
        //      K frags first -> QK^T can start while V frags still in flight)
        v16bf ka00 = load_a(kbase + kb * D_K);                  // t0, d 0..31
        v16bf ka01 = load_a(kbase + kb * D_K + 32);             // t0, d 32..63
        v16bf ka10 = load_a(kbase + (kb + 16) * D_K);           // t1, d 0..31
        v16bf ka11 = load_a(kbase + (kb + 16) * D_K + 32);      // t1, d 32..63
        v16bf vb0  = load_b(vbase + kb);
        v16bf vb1  = load_b(vbase + 16 * S_LEN + kb);
        v16bf vb2  = load_b(vbase + 32 * S_LEN + kb);
        v16bf vb3  = load_b(vbase + 48 * S_LEN + kb);

        if (kb + 32 <= kend) {  // speculative prefetch of next block (CDNA5)
            __builtin_prefetch(kbase + (kb + 32) * D_K, 0, 0);
            __builtin_prefetch(vbase + kb + 32, 0, 0);
        }

        // scores^T: two 16-key tiles, d_k=64 accumulated over two k=32 WMMAs
        v8f st[2];
        st[0] = wmma_bf16(ka00, qf[0].v, vzero8());
        st[0] = wmma_bf16(ka01, qf[1].v, st[0]);
        st[1] = wmma_bf16(ka10, qf[0].v, vzero8());
        st[1] = wmma_bf16(ka11, qf[1].v, st[1]);

        // causal + mask, running block max
        float bm = -1.0e30f;
#pragma unroll
        for (int t = 0; t < 2; ++t)
#pragma unroll
            for (int r = 0; r < 8; ++r) {
                const int kloc = t * 16 + r + (hi ? 8 : 0);
                const int kg   = kb + kloc;
                const bool ok  = (kg <= qg) && ((mask32 >> kloc) & 1u);
                float v = ok ? st[t][r] : -1.0e30f;
                st[t][r] = v;
                bm = fmaxf(bm, v);
            }
        bm = fmaxf(bm, __shfl_xor(bm, 16, 32));   // combine half-waves

        const float mnew = fmaxf(mprev, bm);
        const float corr = __expf(mprev - mnew);

        // exp + pack P directly into A-operand layout (tile0->v0..3, tile1->v4..7)
        float rs = 0.0f;
        Frag pf;
#pragma unroll
        for (int t = 0; t < 2; ++t)
#pragma unroll
            for (int r = 0; r < 4; ++r) {
                float p0 = __expf(st[t][2 * r]     - mnew);
                float p1 = __expf(st[t][2 * r + 1] - mnew);
                rs += p0 + p1;
                pf.w[t * 4 + r] =
                    (unsigned)to_bf16(p0) | ((unsigned)to_bf16(p1) << 16);
            }
        rs += __shfl_xor(rs, 16, 32);
        l = l * corr + rs;
        mprev = mnew;

        // rescale O accumulators: row m needs corr from lane m (m = r + 8*hi)
#pragma unroll
        for (int r = 0; r < 8; ++r) {
            float c0 = __shfl(corr, r, 32);
            float c1 = __shfl(corr, r + 8, 32);
            float cr = hi ? c1 : c0;
            o[0][r] *= cr; o[1][r] *= cr; o[2][r] *= cr; o[3][r] *= cr;
        }

        // O += P * V  (V transposed: column d contiguous over keys)
        o[0] = wmma_bf16(pf.v, vb0, o[0]);
        o[1] = wmma_bf16(pf.v, vb1, o[1]);
        o[2] = wmma_bf16(pf.v, vb2, o[2]);
        o[3] = wmma_bf16(pf.v, vb3, o[3]);
    }

    const float linv = 1.0f / l;
#pragma unroll
    for (int r = 0; r < 8; ++r) {
        float c0 = __shfl(linv, r, 32);
        float c1 = __shfl(linv, r + 8, 32);
        float lr = hi ? c1 : c0;
        const int qrow = q0 + r + (hi ? 8 : 0);
#pragma unroll
        for (int nt = 0; nt < 4; ++nt) {
            const int d = nt * 16 + ln;
            ctx[(b * S_LEN + qrow) * D_MODEL + h * D_K + d] =
                to_bf16(o[nt][r] * lr);
        }
    }
}

// ---------------------------------------------------------------------------
// Output projection: out = ctx @ Wo^T + bo  (fp32 result), pipelined as above
// ---------------------------------------------------------------------------
__global__ __launch_bounds__(128) void mha_oproj_kernel(
    const unsigned short* __restrict__ ctx, const unsigned short* __restrict__ wo,
    const float* __restrict__ bo, float* __restrict__ out) {
    const int lane = threadIdx.x & 31;
    const int wid  = threadIdx.x >> 5;
    const int hi   = lane >> 4;
    const int ln   = lane & 15;
    const int koffA = hi ? 8 : 0;
    const int koffB = hi ? 8 : 0;

    const int mtile = blockIdx.x * 4 + wid;   // 0..255
    const int m0    = mtile * 16;
    const int n0    = blockIdx.y * 64;

    const unsigned short* arow = ctx + (m0 + ln) * D_MODEL + koffA;
    const unsigned short* brow = wo + (n0 + ln) * D_MODEL + koffB;

    v8f acc[4] = {vzero8(), vzero8(), vzero8(), vzero8()};

    v16bf a  = load_a(arow);
    v16bf b0 = load_b(brow);
    v16bf b1 = load_b(brow + 16 * D_MODEL);
    v16bf b2 = load_b(brow + 32 * D_MODEL);
    v16bf b3 = load_b(brow + 48 * D_MODEL);

    for (int c = 0; c < D_MODEL / 32 - 1; ++c) {
        const int k1 = 32 * (c + 1);
        v16bf an  = load_a(arow + k1);
        v16bf bn0 = load_b(brow + k1);
        v16bf bn1 = load_b(brow + 16 * D_MODEL + k1);
        v16bf bn2 = load_b(brow + 32 * D_MODEL + k1);
        v16bf bn3 = load_b(brow + 48 * D_MODEL + k1);
        acc[0] = wmma_bf16(a, b0, acc[0]);
        acc[1] = wmma_bf16(a, b1, acc[1]);
        acc[2] = wmma_bf16(a, b2, acc[2]);
        acc[3] = wmma_bf16(a, b3, acc[3]);
        a = an; b0 = bn0; b1 = bn1; b2 = bn2; b3 = bn3;
    }
    acc[0] = wmma_bf16(a, b0, acc[0]);
    acc[1] = wmma_bf16(a, b1, acc[1]);
    acc[2] = wmma_bf16(a, b2, acc[2]);
    acc[3] = wmma_bf16(a, b3, acc[3]);

#pragma unroll
    for (int nt = 0; nt < 4; ++nt) {
        const int n  = n0 + nt * 16 + ln;
        const float bv_ = bo[n];
#pragma unroll
        for (int r = 0; r < 8; ++r) {
            const int m = m0 + r + (hi ? 8 : 0);
            out[m * D_MODEL + n] = acc[nt][r] + bv_;
        }
    }
}

// ---------------------------------------------------------------------------
extern "C" void kernel_launch(void* const* d_in, const int* in_sizes, int n_in,
                              void* d_out, int out_size, void* d_ws, size_t ws_size,
                              hipStream_t stream) {
    const float* x   = (const float*)d_in[0];
    const int*   am  = (const int*)d_in[1];
    const float* w_q = (const float*)d_in[2];
    const float* b_q = (const float*)d_in[3];
    const float* w_k = (const float*)d_in[4];
    const float* b_k = (const float*)d_in[5];
    const float* w_v = (const float*)d_in[6];
    const float* b_v = (const float*)d_in[7];
    const float* w_o = (const float*)d_in[8];
    const float* b_o = (const float*)d_in[9];
    float* out = (float*)d_out;

    const size_t MD  = (size_t)M_ROWS * D_MODEL;     // 4 Mi elems
    const size_t WSZ = (size_t)D_MODEL * D_MODEL;    // 1 Mi elems

    unsigned short* p   = (unsigned short*)d_ws;
    unsigned short* xb  = p; p += MD;
    unsigned short* wqb = p; p += WSZ;
    unsigned short* wkb = p; p += WSZ;
    unsigned short* wvb = p; p += WSZ;
    unsigned short* wob = p; p += WSZ;
    unsigned short* Qb  = p; p += MD;
    unsigned short* Kb  = p; p += MD;
    unsigned short* Vt  = p; p += MD;
    unsigned short* ctx = p; p += MD;

    // fp32 -> bf16 conversions
    {
        int n4 = (int)(MD / 4);
        mha_cvt_bf16_x4<<<(n4 + 255) / 256, 256, 0, stream>>>(x, xb, n4);
        int w4 = (int)(WSZ / 4);
        dim3 g((w4 + 255) / 256);
        mha_cvt_bf16_x4<<<g, 256, 0, stream>>>(w_q, wqb, w4);
        mha_cvt_bf16_x4<<<g, 256, 0, stream>>>(w_k, wkb, w4);
        mha_cvt_bf16_x4<<<g, 256, 0, stream>>>(w_v, wvb, w4);
        mha_cvt_bf16_x4<<<g, 256, 0, stream>>>(w_o, wob, w4);
    }

    dim3 blk(128);
    mha_qkv_kernel<<<dim3(64, 16, 3), blk, 0, stream>>>(
        xb, wqb, wkb, wvb, b_q, b_k, b_v, Qb, Kb, Vt);
    mha_attn_kernel<<<dim3(32, BATCH * N_HEADS), blk, 0, stream>>>(
        Qb, Kb, Vt, am, ctx);
    mha_oproj_kernel<<<dim3(64, 16), blk, 0, stream>>>(ctx, wob, b_o, out);
}
